// TransformerBlock_15315853377994
// MI455X (gfx1250) — compile-verified, matrix operand order checked by reading, not run on hardware
//
#include <hip/hip_runtime.h>
#include <hip/hip_bf16.h>

typedef __attribute__((ext_vector_type(16))) _Float16 v16h;
typedef __attribute__((ext_vector_type(8)))  _Float16 v8h;
typedef __attribute__((ext_vector_type(8)))  float    v8f;

#define NPIX   16384   // H*W = 128*128
#define IMH    128
#define IMW    128
#define NB     2
#define DIMC   192
#define HEADS  4
#define CHN    48
#define HIDC   510

// ---------------------------------------------------------------------------
// Tiled B-fragment layout helper.
// A (32k x 16n) tile holds 512 halves ordered so that lane L of a wave reads
// its WMMA B fragment as two contiguous 16-byte chunks:
//   chunk0 (k = 8*(L>>4)+e)    at tile + L*8 + e
//   chunk1 (k = 16+8*(L>>4)+e) at tile + 256 + L*8 + e
// ---------------------------------------------------------------------------
__device__ __forceinline__ size_t btile_off(int tk, int tn, int kk, int nn, int NT) {
  int chunk = kk >> 4;
  int hs    = (kk >> 3) & 1;
  int e     = kk & 7;
  int lane  = nn + (hs << 4);
  return (((size_t)tk * NT + tn) << 9) + (chunk << 8) + (lane << 3) + e;
}

// ---------------------------------------------------------------------------
// LayerNorm over channel dim (per b, per pixel)
// ---------------------------------------------------------------------------
__global__ void layernorm_k(const float* __restrict__ x, const float* __restrict__ w,
                            const float* __restrict__ bias, float* __restrict__ y) {
  int idx = blockIdx.x * 256 + threadIdx.x;
  if (idx >= NB * NPIX) return;
  int b = idx >> 14;
  int n = idx & (NPIX - 1);
  const float* xp = x + (size_t)b * DIMC * NPIX + n;
  float mu = 0.f;
  for (int c = 0; c < DIMC; ++c) mu += xp[(size_t)c * NPIX];
  mu *= (1.0f / DIMC);
  float var = 0.f;
  for (int c = 0; c < DIMC; ++c) { float d = xp[(size_t)c * NPIX] - mu; var += d * d; }
  var *= (1.0f / DIMC);
  float inv = rsqrtf(var + 1e-5f);
  float* yp = y + (size_t)b * DIMC * NPIX + n;
  for (int c = 0; c < DIMC; ++c)
    yp[(size_t)c * NPIX] = (xp[(size_t)c * NPIX] - mu) * inv * w[c] + bias[c];
}

// ---------------------------------------------------------------------------
// f32 [b][K][N] -> f16 WMMA-B-tiled [b][Kpad/32][N/16][512] (zero pad K rows)
// ---------------------------------------------------------------------------
__global__ void cvt_x16t(const float* __restrict__ src, _Float16* __restrict__ dst,
                         int K, int Kpad) {
  size_t idx = (size_t)blockIdx.x * 256 + threadIdx.x;
  size_t total = (size_t)NB * Kpad * NPIX;
  if (idx >= total) return;
  int n = (int)(idx & (NPIX - 1));
  size_t t2 = idx >> 14;
  int k = (int)(t2 % Kpad);
  int b = (int)(t2 / Kpad);
  _Float16 v = (_Float16)0.f;
  if (k < K) v = (_Float16)src[((size_t)b * K + k) * NPIX + n];
  size_t base = (size_t)b * Kpad * NPIX;
  dst[base + btile_off(k >> 5, n >> 4, k & 31, n & 15, NPIX >> 4)] = v;
}

// f32 weights [M][K] -> f16 [Mpad][Kpad] zero padded (A fragments are row-major)
__global__ void cvt_w16(const float* __restrict__ src, _Float16* __restrict__ dst,
                        int M, int K, int Mpad, int Kpad) {
  int idx = blockIdx.x * 256 + threadIdx.x;
  if (idx >= Mpad * Kpad) return;
  int k = idx % Kpad, m = idx / Kpad;
  dst[idx] = (m < M && k < K) ? (_Float16)src[(size_t)m * K + k] : (_Float16)0.f;
}

// ---------------------------------------------------------------------------
// WMMA GEMM: out[b][m][n] = sum_k W[m][k] * X[b][k][n] (+ residual)
// W:  f16 [Mpad x K] row-major (K multiple of 32, rows >= M zeroed)
// XT: f16 tiled-B layout per batch; out: f32 [b][M][N]
// block = 256 thr (8 waves); wave -> one 16x16 tile; grid (N/128, Mpad/16, B)
// Inner loop: 2x b128 A loads + 2x b128 B loads + 1 WMMA.
// ---------------------------------------------------------------------------
__global__ void gemm_wmma(const _Float16* __restrict__ W16, const _Float16* __restrict__ XT,
                          float* __restrict__ out, const float* __restrict__ res,
                          int M, int K, int N) {
  int lane = threadIdx.x & 31, wave = threadIdx.x >> 5;
  int lrow = lane & 15, hs = lane >> 4;
  int n0 = (blockIdx.x * 8 + wave) * 16;
  int m0 = blockIdx.y * 16;
  int b  = blockIdx.z;
  int NT = N >> 4;
  const _Float16* ap0 = W16 + (size_t)(m0 + lrow) * K + 8 * hs;
  const _Float16* bt0 = XT + (size_t)b * K * N + ((size_t)(n0 >> 4) << 9) + (lane << 3);
  const size_t bstep = (size_t)NT << 9;   // one k-tile forward
  v8f acc = {};
  for (int k0 = 0; k0 < K; k0 += 32) {
    v16h a, bf;
    const _Float16* ap = ap0 + k0;
    v8h alo = *(const v8h*)ap;
    v8h ahi = *(const v8h*)(ap + 16);
    const _Float16* bp = bt0 + (size_t)(k0 >> 5) * bstep;
    v8h blo = *(const v8h*)bp;
    v8h bhi = *(const v8h*)(bp + 256);
#pragma unroll
    for (int e = 0; e < 8; ++e) {
      a[e] = alo[e]; a[8 + e] = ahi[e];
      bf[e] = blo[e]; bf[8 + e] = bhi[e];
    }
    __builtin_prefetch((const void*)(bp + bstep), 0, 1);
    acc = __builtin_amdgcn_wmma_f32_16x16x32_f16(false, a, false, bf, (short)0, acc,
                                                 false, false);
  }
#pragma unroll
  for (int r = 0; r < 8; ++r) {
    int m = m0 + 8 * hs + r;
    if (m < M) {
      size_t o = ((size_t)b * M + m) * (size_t)N + n0 + lrow;
      float v = acc[r];
      if (res) v += res[o];
      out[o] = v;
    }
  }
}

// ---------------------------------------------------------------------------
// Depthwise conv, zero pad, optional relu. w: [C][(2R+1)^2]
// ---------------------------------------------------------------------------
__global__ void dwconv_k(const float* __restrict__ in, const float* __restrict__ w,
                         float* __restrict__ out, int C, int R, int doRelu) {
  size_t idx = (size_t)blockIdx.x * 256 + threadIdx.x;
  size_t total = (size_t)NB * C * NPIX;
  if (idx >= total) return;
  int n = (int)(idx & (NPIX - 1));
  size_t bc = idx >> 14;
  int c = (int)(bc % C);
  int y = n >> 7, x = n & (IMW - 1);
  const float* ip = in + bc * NPIX;
  int kw = 2 * R + 1;
  const float* wp = w + (size_t)c * kw * kw;
  float acc = 0.f;
  for (int dy = -R; dy <= R; ++dy) {
    int yy = y + dy; if (yy < 0 || yy >= IMH) continue;
    for (int dx = -R; dx <= R; ++dx) {
      int xx = x + dx; if (xx < 0 || xx >= IMW) continue;
      acc += ip[yy * IMW + xx] * wp[(dy + R) * kw + (dx + R)];
    }
  }
  if (doRelu) acc = fmaxf(acc, 0.f);
  out[idx] = acc;
}

// ---------------------------------------------------------------------------
// Grouped conv: 510 groups, 2 in-ch -> 1 out-ch, relu.
// Input is virtual concat: ch<510 -> x3half[ch], ch>=510 -> x5half[ch-510]
// ---------------------------------------------------------------------------
__global__ void grouped2_k(const float* __restrict__ x3, const float* __restrict__ x5,
                           const float* __restrict__ w, float* __restrict__ out,
                           int half, int R) {
  size_t idx = (size_t)blockIdx.x * 256 + threadIdx.x;
  size_t total = (size_t)NB * HIDC * NPIX;
  if (idx >= total) return;
  int n = (int)(idx & (NPIX - 1));
  size_t t2 = idx >> 14;
  int g = (int)(t2 % HIDC);
  int b = (int)(t2 / HIDC);
  int y = n >> 7, x = n & (IMW - 1);
  const float* srcA = x3 + ((size_t)b * 2 * HIDC + half * HIDC) * NPIX;
  const float* srcB = x5 + ((size_t)b * 2 * HIDC + half * HIDC) * NPIX;
  int kw = 2 * R + 1;
  float acc = 0.f;
  for (int i = 0; i < 2; ++i) {
    int cc = 2 * g + i;
    const float* ip = (cc < HIDC) ? (srcA + (size_t)cc * NPIX)
                                  : (srcB + (size_t)(cc - HIDC) * NPIX);
    const float* wp = w + ((size_t)g * 2 + i) * kw * kw;
    for (int dy = -R; dy <= R; ++dy) {
      int yy = y + dy; if (yy < 0 || yy >= IMH) continue;
      for (int dx = -R; dx <= R; ++dx) {
        int xx = x + dx; if (xx < 0 || xx >= IMW) continue;
        acc += ip[yy * IMW + xx] * wp[(dy + R) * kw + (dx + R)];
      }
    }
  }
  acc = fmaxf(acc, 0.f);
  out[((size_t)b * 2 * HIDC + half * HIDC + g) * NPIX + n] = acc;
}

// ---------------------------------------------------------------------------
// L2-normalize q,k rows (channels 0..383) over N, convert to f16 row-major.
// grid = NB*384 blocks, 256 threads
// ---------------------------------------------------------------------------
__global__ void l2norm_cvt(const float* __restrict__ src, _Float16* __restrict__ dst) {
  __shared__ float red[256];
  int bc = blockIdx.x;                  // (b, c<384)
  int b = bc / (2 * DIMC);
  int c = bc % (2 * DIMC);
  const float* row = src + ((size_t)b * 3 * DIMC + c) * NPIX;
  _Float16* orow = dst + ((size_t)b * 3 * DIMC + c) * NPIX;
  float s = 0.f;
  for (int i = threadIdx.x; i < NPIX; i += 256) { float v = row[i]; s += v * v; }
  red[threadIdx.x] = s;
  __syncthreads();
  for (int off = 128; off > 0; off >>= 1) {
    if ((int)threadIdx.x < off) red[threadIdx.x] += red[threadIdx.x + off];
    __syncthreads();
  }
  float scale = 1.0f / fmaxf(sqrtf(red[0]), 1e-12f);
  for (int i = threadIdx.x; i < NPIX; i += 256) orow[i] = (_Float16)(row[i] * scale);
}

// ---------------------------------------------------------------------------
// Stage v (dwqkv channels 384..575) into tiled-B layout per (b,h):
// VT[bh] : Kpad=64 x N=16384, pad rows zeroed.
// ---------------------------------------------------------------------------
__global__ void stage_v_tiled(const float* __restrict__ dwqkv, _Float16* __restrict__ VT) {
  size_t idx = (size_t)blockIdx.x * 256 + threadIdx.x;
  size_t total = (size_t)8 * 64 * NPIX;
  if (idx >= total) return;
  int n = (int)(idx & (NPIX - 1));
  size_t t2 = idx >> 14;
  int k = (int)(t2 % 64);
  int bh = (int)(t2 / 64);
  int b = bh >> 2, h = bh & 3;
  _Float16 v = (_Float16)0.f;
  if (k < CHN)
    v = (_Float16)dwqkv[((size_t)b * 3 * DIMC + 2 * DIMC + h * CHN + k) * NPIX + n];
  VT[(size_t)bh * 64 * NPIX + btile_off(k >> 5, n >> 4, k & 31, n & 15, NPIX >> 4)] = v;
}

// ---------------------------------------------------------------------------
// attn[bh][c][d] = temperature[h] * sum_n q[c][n]*k[d][n]   (48x48, K=16384)
// Both fragments load contiguously. grid (9 tiles, 8 bh), block = 32
// ---------------------------------------------------------------------------
__global__ void qk_attn(const _Float16* __restrict__ qkv16, const float* __restrict__ temp,
                        float* __restrict__ attn) {
  int lane = threadIdx.x;
  int lrow = lane & 15, hs = lane >> 4;
  int tile = blockIdx.x, mt = tile / 3, nt = tile % 3;
  int bh = blockIdx.y, b = bh >> 2, h = bh & 3;
  const _Float16* Q  = qkv16 + ((size_t)b * 3 * DIMC + h * CHN) * NPIX;
  const _Float16* Kp = qkv16 + ((size_t)b * 3 * DIMC + DIMC + h * CHN) * NPIX;
  const _Float16* ap0 = Q  + (size_t)(mt * 16 + lrow) * NPIX + 8 * hs;
  const _Float16* bp0 = Kp + (size_t)(nt * 16 + lrow) * NPIX + 8 * hs;
  v8f acc = {};
  for (int k0 = 0; k0 < NPIX; k0 += 32) {
    v16h a, bf;
    v8h alo = *(const v8h*)(ap0 + k0);
    v8h ahi = *(const v8h*)(ap0 + k0 + 16);
    v8h blo = *(const v8h*)(bp0 + k0);
    v8h bhi = *(const v8h*)(bp0 + k0 + 16);
#pragma unroll
    for (int e = 0; e < 8; ++e) {
      a[e] = alo[e]; a[8 + e] = ahi[e];
      bf[e] = blo[e]; bf[8 + e] = bhi[e];
    }
    acc = __builtin_amdgcn_wmma_f32_16x16x32_f16(false, a, false, bf, (short)0, acc,
                                                 false, false);
  }
  float ts = temp[h];
#pragma unroll
  for (int r = 0; r < 8; ++r) {
    int m = mt * 16 + 8 * hs + r;
    int n = nt * 16 + lrow;
    attn[((size_t)bh * CHN + m) * CHN + n] = acc[r] * ts;
  }
}

// ---------------------------------------------------------------------------
// Exact 48x48 FFT band-reject filter (rolls folded into shifted mask):
// F = fft2(attn)/2304 ; attnC = sum_i a_i * |ifft2(F * keep_{r_i})|
// keep(v,u)=0 iff (v+24)%48 and (u+24)%48 both inside [24-r,24+r)
// ---------------------------------------------------------------------------
__global__ void fft_filter(const float* __restrict__ attn, float* __restrict__ attnC,
                           const float* a1, const float* a2, const float* a3,
                           const float* a4) {
  __shared__ float Fr[2304], Fi[2304], Tr[2304], Ti[2304], Acc[2304];
  const float TWO_PI = 6.28318530717958647692f;
  int bh = blockIdx.x;
  int t = threadIdx.x;  // 64 threads
  for (int i = t; i < 2304; i += 64) {
    Tr[i] = attn[(size_t)bh * 2304 + i]; Ti[i] = 0.f; Acc[i] = 0.f;
  }
  __syncthreads();
  // forward rows
  for (int i = t; i < 2304; i += 64) {
    int y = i / 48, u = i % 48; float sr = 0.f, si = 0.f;
    for (int x = 0; x < 48; ++x) {
      float ang = -TWO_PI * (float)((u * x) % 48) * (1.0f / 48.f);
      float s, c; sincosf(ang, &s, &c);
      float vr = Tr[y * 48 + x], vi = Ti[y * 48 + x];
      sr += vr * c - vi * s; si += vr * s + vi * c;
    }
    Fr[i] = sr; Fi[i] = si;
  }
  __syncthreads();
  // forward cols + 1/2304 scale -> Tr/Ti hold F
  for (int i = t; i < 2304; i += 64) {
    int v = i / 48, u = i % 48; float sr = 0.f, si = 0.f;
    for (int y = 0; y < 48; ++y) {
      float ang = -TWO_PI * (float)((v * y) % 48) * (1.0f / 48.f);
      float s, c; sincosf(ang, &s, &c);
      float vr = Fr[y * 48 + u], vi = Fi[y * 48 + u];
      sr += vr * c - vi * s; si += vr * s + vi * c;
    }
    Tr[i] = sr * (1.f / 2304.f); Ti[i] = si * (1.f / 2304.f);
  }
  __syncthreads();
  const int rad[4] = {4, 9, 14, 19};
  float coef[4] = {a1[0], a2[0], a3[0], a4[0]};
  for (int br = 0; br < 4; ++br) {
    int r = rad[br];
    // masked inverse cols
    for (int i = t; i < 2304; i += 64) {
      int y = i / 48, u = i % 48; float sr = 0.f, si = 0.f;
      int us = (u + 24) % 48;
      bool uin = (us >= 24 - r) && (us < 24 + r);
      for (int v = 0; v < 48; ++v) {
        int vs = (v + 24) % 48;
        bool vin = (vs >= 24 - r) && (vs < 24 + r);
        if (uin && vin) continue;
        float ang = TWO_PI * (float)((v * y) % 48) * (1.0f / 48.f);
        float s, c; sincosf(ang, &s, &c);
        float vr = Tr[v * 48 + u], vi = Ti[v * 48 + u];
        sr += vr * c - vi * s; si += vr * s + vi * c;
      }
      Fr[i] = sr; Fi[i] = si;
    }
    __syncthreads();
    // inverse rows + abs accumulate
    for (int i = t; i < 2304; i += 64) {
      int y = i / 48, x = i % 48; float sr = 0.f, si = 0.f;
      for (int u = 0; u < 48; ++u) {
        float ang = TWO_PI * (float)((u * x) % 48) * (1.0f / 48.f);
        float s, c; sincosf(ang, &s, &c);
        float vr = Fr[y * 48 + u], vi = Fi[y * 48 + u];
        sr += vr * c - vi * s; si += vr * s + vi * c;
      }
      Acc[i] += coef[br] * sqrtf(sr * sr + si * si);
    }
    __syncthreads();
  }
  for (int i = t; i < 2304; i += 64) attnC[(size_t)bh * 2304 + i] = Acc[i];
}

// attnC f32 [8][48][48] -> f16 [8][48][64] zero-padded K (A-fragment row-major)
__global__ void cvt_attnc(const float* __restrict__ attnC, _Float16* __restrict__ a16) {
  int idx = blockIdx.x * 256 + threadIdx.x;
  if (idx >= 8 * 48 * 64) return;
  int k = idx % 64;
  int m = (idx / 64) % 48;
  int bh = idx / (64 * 48);
  a16[idx] = (k < 48) ? (_Float16)attnC[((size_t)bh * 48 + m) * 48 + k] : (_Float16)0.f;
}

// ---------------------------------------------------------------------------
// out[b][h*48+m][n] = sum_d attnC[bh][m][d] * v[bh][d][n]   (M=48,Kpad=64,N=16384)
// V from tiled-B staging (pads zeroed). grid (128, 3, 8), block 256 (8 waves)
// ---------------------------------------------------------------------------
__global__ void av_gemm(const _Float16* __restrict__ A16, const _Float16* __restrict__ VT,
                        float* __restrict__ out) {
  int lane = threadIdx.x & 31, wave = threadIdx.x >> 5;
  int lrow = lane & 15, hs = lane >> 4;
  int n0 = (blockIdx.x * 8 + wave) * 16;
  int mt = blockIdx.y;
  int bh = blockIdx.z, b = bh >> 2, h = bh & 3;
  const _Float16* A = A16 + (size_t)bh * 48 * 64;
  const _Float16* bt0 = VT + (size_t)bh * 64 * NPIX + ((size_t)(n0 >> 4) << 9) + (lane << 3);
  const size_t bstep = (size_t)(NPIX >> 4) << 9;
  int ncol = n0 + lrow;
  v8f acc = {};
#pragma unroll
  for (int k0 = 0; k0 < 64; k0 += 32) {
    v16h a, bf;
    const _Float16* ap = A + (size_t)(mt * 16 + lrow) * 64 + k0 + 8 * hs;
    const _Float16* bp = bt0 + (size_t)(k0 >> 5) * bstep;
    v8h blo = *(const v8h*)bp;
    v8h bhi = *(const v8h*)(bp + 256);
#pragma unroll
    for (int e = 0; e < 8; ++e) {
      a[e] = ap[e]; a[8 + e] = ap[16 + e];
      bf[e] = blo[e]; bf[8 + e] = bhi[e];
    }
    acc = __builtin_amdgcn_wmma_f32_16x16x32_f16(false, a, false, bf, (short)0, acc,
                                                 false, false);
  }
#pragma unroll
  for (int r = 0; r < 8; ++r) {
    int m = mt * 16 + 8 * hs + r;
    out[((size_t)b * DIMC + h * CHN + m) * NPIX + ncol] = acc[r];
  }
}

// ---------------------------------------------------------------------------
// Launcher
// ---------------------------------------------------------------------------
extern "C" void kernel_launch(void* const* d_in, const int* in_sizes, int n_in,
                              void* d_out, int out_size, void* d_ws, size_t ws_size,
                              hipStream_t stream) {
  const float* x       = (const float*)d_in[0];
  const float* ln1_w   = (const float*)d_in[1];
  const float* ln1_b   = (const float*)d_in[2];
  const float* ln2_w   = (const float*)d_in[3];
  const float* ln2_b   = (const float*)d_in[4];
  const float* qkv_w   = (const float*)d_in[5];
  const float* qkvdw_w = (const float*)d_in[6];
  const float* proj_w  = (const float*)d_in[7];
  const float* temp    = (const float*)d_in[8];
  const float* a1      = (const float*)d_in[9];
  const float* a2      = (const float*)d_in[10];
  const float* a3      = (const float*)d_in[11];
  const float* a4      = (const float*)d_in[12];
  const float* pin_w   = (const float*)d_in[13];
  const float* dw3_w   = (const float*)d_in[14];
  const float* dw5_w   = (const float*)d_in[15];
  const float* dw31_w  = (const float*)d_in[16];
  const float* dw51_w  = (const float*)d_in[17];
  const float* pout_w  = (const float*)d_in[18];
  float* out = (float*)d_out;

  char* wsp = (char*)d_ws;
  const size_t szBig = (size_t)NB * 1020 * NPIX * sizeof(float);   // 133.7 MB
  float* R0 = (float*)(wsp);
  float* R1 = (float*)(wsp + szBig);
  float* R2 = (float*)(wsp + 2 * szBig);
  float* R3 = (float*)(wsp + 3 * szBig);                            // LN out / 25 MB
  size_t off = 3 * szBig + (size_t)NB * DIMC * NPIX * sizeof(float);
  _Float16* XH = (_Float16*)(wsp + off);                            // tiled acts [b][1024][N]
  off += (size_t)NB * 1024 * NPIX * sizeof(_Float16);
  _Float16* QKV16 = (_Float16*)(wsp + off);                         // f16 q,k rows [b][576][N]
  off += (size_t)NB * 576 * NPIX * sizeof(_Float16);
  _Float16* VT = (_Float16*)(wsp + off);                            // tiled v [bh][64][N]
  off += (size_t)8 * 64 * NPIX * sizeof(_Float16);
  _Float16* Wqkv  = (_Float16*)(wsp + off); off += (size_t)576  * 192  * 2;
  _Float16* Wproj = (_Float16*)(wsp + off); off += (size_t)192  * 192  * 2;
  _Float16* Wpin  = (_Float16*)(wsp + off); off += (size_t)1024 * 192  * 2;
  _Float16* Wpout = (_Float16*)(wsp + off); off += (size_t)192  * 1024 * 2;
  _Float16* AC16  = (_Float16*)(wsp + off); off += (size_t)8 * 48 * 64 * 2;
  float* ATTN  = (float*)(wsp + off); off += (size_t)8 * 48 * 48 * 4;
  float* ATTNC = (float*)(wsp + off);

  auto blocks = [](size_t n) { return (unsigned)((n + 255) / 256); };

  // ---- attention branch ----
  layernorm_k<<<blocks((size_t)NB * NPIX), 256, 0, stream>>>(x, ln1_w, ln1_b, R3);
  cvt_x16t<<<blocks((size_t)NB * 192 * NPIX), 256, 0, stream>>>(R3, XH, 192, 192);
  cvt_w16<<<blocks((size_t)576 * 192), 256, 0, stream>>>(qkv_w, Wqkv, 576, 192, 576, 192);
  gemm_wmma<<<dim3(NPIX / 128, 36, NB), 256, 0, stream>>>(Wqkv, XH, R0, nullptr,
                                                          576, 192, NPIX);
  dwconv_k<<<blocks((size_t)NB * 576 * NPIX), 256, 0, stream>>>(R0, qkvdw_w, R1, 576, 1, 0);
  l2norm_cvt<<<NB * 384, 256, 0, stream>>>(R1, QKV16);
  stage_v_tiled<<<blocks((size_t)8 * 64 * NPIX), 256, 0, stream>>>(R1, VT);
  qk_attn<<<dim3(9, 8), 32, 0, stream>>>(QKV16, temp, ATTN);
  fft_filter<<<8, 64, 0, stream>>>(ATTN, ATTNC, a1, a2, a3, a4);
  cvt_attnc<<<blocks((size_t)8 * 48 * 64), 256, 0, stream>>>(ATTNC, AC16);
  av_gemm<<<dim3(NPIX / 128, 3, 8), 256, 0, stream>>>(AC16, VT, R1);
  cvt_x16t<<<blocks((size_t)NB * 192 * NPIX), 256, 0, stream>>>(R1, XH, 192, 192);
  cvt_w16<<<blocks((size_t)192 * 192), 256, 0, stream>>>(proj_w, Wproj, 192, 192, 192, 192);
  // d_out = x + proj(attn_out)
  gemm_wmma<<<dim3(NPIX / 128, 12, NB), 256, 0, stream>>>(Wproj, XH, out, x,
                                                          192, 192, NPIX);

  // ---- FFN branch ----
  layernorm_k<<<blocks((size_t)NB * NPIX), 256, 0, stream>>>(out, ln2_w, ln2_b, R3);
  cvt_x16t<<<blocks((size_t)NB * 192 * NPIX), 256, 0, stream>>>(R3, XH, 192, 192);
  cvt_w16<<<blocks((size_t)1024 * 192), 256, 0, stream>>>(pin_w, Wpin, 1020, 192, 1024, 192);
  gemm_wmma<<<dim3(NPIX / 128, 64, NB), 256, 0, stream>>>(Wpin, XH, R0, nullptr,
                                                          1020, 192, NPIX);
  dwconv_k<<<blocks((size_t)NB * 1020 * NPIX), 256, 0, stream>>>(R0, dw3_w, R1, 1020, 1, 1);
  dwconv_k<<<blocks((size_t)NB * 1020 * NPIX), 256, 0, stream>>>(R0, dw5_w, R2, 1020, 2, 1);
  grouped2_k<<<blocks((size_t)NB * HIDC * NPIX), 256, 0, stream>>>(R1, R2, dw31_w, R0, 0, 1);
  grouped2_k<<<blocks((size_t)NB * HIDC * NPIX), 256, 0, stream>>>(R1, R2, dw51_w, R0, 1, 2);
  cvt_x16t<<<blocks((size_t)NB * 1024 * NPIX), 256, 0, stream>>>(R0, XH, 1020, 1024);
  cvt_w16<<<blocks((size_t)192 * 1024), 256, 0, stream>>>(pout_w, Wpout, 192, 1020, 192, 1024);
  // d_out += pout(ffn)
  gemm_wmma<<<dim3(NPIX / 128, 12, NB), 256, 0, stream>>>(Wpout, XH, out, out,
                                                          192, 1024, NPIX);
}
